// ExtractionAndMarkovTemplateMatching_54425825575668
// MI455X (gfx1250) — matrix operand order repaired and checked by reading, not compiled, not run on hardware
//
#include <hip/hip_runtime.h>
#include <hip/hip_bf16.h>
#include <math.h>

#define Bv 32
#define Tv 2048
#define Gv 32000
#define Ev 256
#define Mv 16
#define Sv 32

typedef __attribute__((ext_vector_type(2)))  float  v2f;
typedef __attribute__((ext_vector_type(8)))  float  v8f;
typedef __attribute__((ext_vector_type(16))) __bf16 v16bf;

union BF16x16 { v16bf v; uint4 q[2]; unsigned short u[16]; };

__device__ __forceinline__ unsigned short f2bf(float f) {
  unsigned u = __float_as_uint(f);
  u += 0x7FFFu + ((u >> 16) & 1u);           // round-to-nearest-even
  return (unsigned short)(u >> 16);
}

// ---- K0: logT (transposed) and z0 row ---------------------------------------
__global__ void k_prep(const float* __restrict__ trans, const float* __restrict__ initd,
                       float* __restrict__ logTt, float* __restrict__ z0row) {
  int tid = threadIdx.x;                      // 512 threads
  int m = tid >> 5, i = tid & 31;
  const float* row = trans + (size_t)(m * Sv + i) * Sv;
  float mx = -INFINITY;
  for (int s2 = 0; s2 < Sv; ++s2) { float v = row[s2] + (s2 == i ? 5.f : 0.f); mx = fmaxf(mx, v); }
  float sm = 0.f;
  for (int s2 = 0; s2 < Sv; ++s2) { float v = row[s2] + (s2 == i ? 5.f : 0.f); sm += __expf(v - mx); }
  float lse = mx + __logf(sm);
  for (int s2 = 0; s2 < Sv; ++s2) {
    float v = row[s2] + (s2 == i ? 5.f : 0.f);
    logTt[(size_t)(m * Sv + s2) * Sv + i] = v - lse;   // [m][s2][s1]
  }
  const float* ir = initd + m * Sv;
  float mx2 = -INFINITY; for (int j = 0; j < Sv; ++j) mx2 = fmaxf(mx2, ir[j]);
  float sm2 = 0.f;       for (int j = 0; j < Sv; ++j) sm2 += __expf(ir[j] - mx2);
  z0row[m * Sv + i] = ir[i] - (mx2 + __logf(sm2));
}

// ---- K1: embedding gather ---------------------------------------------------
__global__ void k_gather(const int* __restrict__ x, const float* __restrict__ embW,
                         float* __restrict__ xx) {
  int bt = blockIdx.x;
  int g  = x[bt];
  const float4* src = (const float4*)(embW + (size_t)g * Ev);
  float4*       dst = (float4*)(xx + (size_t)bt * Ev);
  dst[threadIdx.x] = src[threadIdx.x];        // 64 threads * float4 = 256 floats
}

// ---- K2: softmax-over-T stats for attention ---------------------------------
__global__ void k_att_stats(const float* __restrict__ xx,
                            float* __restrict__ amax, float* __restrict__ asum) {
  __shared__ float red[256];
  int row = blockIdx.x; int b = row >> 5, s = row & 31;
  int tid = threadIdx.x;
  float mx = -INFINITY;
  for (int t = tid; t < Tv; t += 256) mx = fmaxf(mx, xx[(size_t)(b * Tv + t) * Ev + s]);
  red[tid] = mx; __syncthreads();
  for (int off = 128; off > 0; off >>= 1) { if (tid < off) red[tid] = fmaxf(red[tid], red[tid + off]); __syncthreads(); }
  mx = red[0]; __syncthreads();
  float sm = 0.f;
  for (int t = tid; t < Tv; t += 256) sm += __expf(xx[(size_t)(b * Tv + t) * Ev + s] - mx);
  red[tid] = sm; __syncthreads();
  for (int off = 128; off > 0; off >>= 1) { if (tid < off) red[tid] += red[tid + off]; __syncthreads(); }
  if (tid == 0) { amax[row] = mx; asum[row] = red[0]; }
}

// ---- K3: vocab_W fp32 -> bf16 ----------------------------------------------
__global__ void k_cvt_vocab(const float* __restrict__ w, unsigned short* __restrict__ wb, int n) {
  int i = (blockIdx.x * blockDim.x + threadIdx.x) * 4;
  if (i + 3 < n) {
    wb[i]     = f2bf(w[i]);     wb[i + 1] = f2bf(w[i + 1]);
    wb[i + 2] = f2bf(w[i + 2]); wb[i + 3] = f2bf(w[i + 3]);
  }
}

// ---- K4: states = att^T @ xx via f32 WMMA 16x16x4 ---------------------------
__global__ void k_states(const float* __restrict__ xx, const float* __restrict__ amax,
                         const float* __restrict__ asum, float* __restrict__ st,
                         unsigned short* __restrict__ stbf) {
  int wid   = blockIdx.x * 4 + (threadIdx.x >> 5);  // 1024 waves
  int lane  = threadIdx.x & 31;
  int etile = wid & 15, stile = (wid >> 4) & 1, b = wid >> 5;
  int mi = lane & 15, hi = lane >> 4;
  int s  = stile * 16 + mi;                          // A M-row (state index)
  int eg = etile * 16 + mi;                          // B N-col (embed index)
  float am  = amax[b * Sv + s];
  float inv = 1.f / asum[b * Sv + s];
  const float* base = xx + (size_t)b * Tv * Ev;
  v8f c = {};
  for (int k0 = 0; k0 < Tv; k0 += 4) {
    int t0 = k0 + 2 * hi;                            // f32 WMMA A/B: k = 2*hi + j
    v2f a, bm;
    a[0]  = __expf(base[(size_t)t0 * Ev + s] - am) * inv;
    a[1]  = __expf(base[(size_t)(t0 + 1) * Ev + s] - am) * inv;
    bm[0] = base[(size_t)t0 * Ev + eg];
    bm[1] = base[(size_t)(t0 + 1) * Ev + eg];
    c = __builtin_amdgcn_wmma_f32_16x16x4_f32(false, a, false, bm, (short)0, c, false, false);
  }
#pragma unroll
  for (int r = 0; r < 8; ++r) {                      // D: lane n=mi, rows m=r+8*hi
    int so = stile * 16 + r + 8 * hi;
    float v = c[r];
    st  [(size_t)(b * Sv + so) * Ev + eg] = v;
    stbf[(size_t)(b * Sv + so) * Ev + eg] = f2bf(v);
  }
}

// ---- K5: logits = states @ vocab_W^T + b via bf16 WMMA 16x16x32 -------------
// 2 waves/block; each wave owns one (mtile, 25-ntile chunk). Vocab tiles are
// staged global->LDS with async-to-LDS B128 copies, double-buffered on ASYNCcnt.
#define ROWB ((Ev + 8) * 2)                 /* padded LDS row stride in bytes: 528 */
__global__ void k_logits(const unsigned short* __restrict__ stbf,
                         const unsigned short* __restrict__ vwb,
                         const float* __restrict__ vb,
                         float* __restrict__ logits) {
  __shared__ __align__(16) char btile[2][2][16 * ROWB];   // [wave][buf][16 rows]
  int w    = threadIdx.x >> 5;
  int wid  = blockIdx.x * 2 + w;                          // 0..5119
  int lane = threadIdx.x & 31;
  int mtile = wid / 80, chunk = wid % 80;
  int mi = lane & 15, hi = lane >> 4;
  int row = mtile * 16 + mi;

  // Preload A (all 8 K-slices) as B128 loads; reused across 25 N-tiles.
  BF16x16 A[8];
#pragma unroll
  for (int ks = 0; ks < 8; ++ks) {
    const unsigned short* ap = stbf + (size_t)row * Ev + ks * 32 + hi * 8;
    A[ks].q[0] = *(const uint4*)(ap);
    A[ks].q[1] = *(const uint4*)(ap + 16);
  }

  // Async-stage one 16x256 bf16 vocab tile (16B granules) into LDS buffer.
  auto stage = [&](int ntile, int buf) {
    unsigned base = (unsigned)(uintptr_t)(&btile[w][buf][0]);
#pragma unroll
    for (int it = 0; it < 16; ++it) {
      int idx = lane + 32 * it;                 // 0..511 sixteen-byte chunks
      int r   = idx >> 5;                       // tile row 0..15
      int cb  = (idx & 31) * 16;                // byte offset within row
      const char* gp = (const char*)(vwb + (size_t)(ntile * 16 + r) * Ev) + cb;
      unsigned loff = base + (unsigned)(r * ROWB + cb);
      asm volatile("global_load_async_to_lds_b128 %0, %1, off"
                   :: "v"(loff), "v"(gp) : "memory");
    }
  };

  stage(chunk * 25, 0);
  for (int nt = 0; nt < 25; ++nt) {
    int ntile = chunk * 25 + nt;
    int buf = nt & 1;
    if (nt + 1 < 25) {
      stage(ntile + 1, buf ^ 1);
      asm volatile("s_wait_asynccnt 0x10" ::: "memory");  // tile nt complete
    } else {
      asm volatile("s_wait_asynccnt 0x0" ::: "memory");
    }
    int g = ntile * 16 + mi;
    v8f c = {};
#pragma unroll
    for (int ks = 0; ks < 8; ++ks) {
      BF16x16 Bm;
      const char* bp = &btile[w][buf][mi * ROWB + (ks * 32 + hi * 8) * 2];
      Bm.q[0] = *(const uint4*)(bp);
      Bm.q[1] = *(const uint4*)(bp + 32);
      c = __builtin_amdgcn_wmma_f32_16x16x32_bf16(false, A[ks].v, false, Bm.v, (short)0, c, false, false);
    }
    float bias = vb[g];
#pragma unroll
    for (int r = 0; r < 8; ++r) {
      int mrow = mtile * 16 + r + 8 * hi;
      logits[(size_t)mrow * Gv + g] = c[r] + bias;
    }
  }
}

// ---- K6: per-row logsumexp over G -------------------------------------------
__global__ void k_rowlse(const float* __restrict__ logits, float* __restrict__ lse) {
  __shared__ float red[256];
  int row = blockIdx.x, tid = threadIdx.x;
  const float* p = logits + (size_t)row * Gv;
  float mx = -INFINITY;
  for (int g = tid; g < Gv; g += 256) mx = fmaxf(mx, p[g]);
  red[tid] = mx; __syncthreads();
  for (int off = 128; off > 0; off >>= 1) { if (tid < off) red[tid] = fmaxf(red[tid], red[tid + off]); __syncthreads(); }
  mx = red[0]; __syncthreads();
  float sm = 0.f;
  for (int g = tid; g < Gv; g += 256) sm += __expf(p[g] - mx);
  red[tid] = sm; __syncthreads();
  for (int off = 128; off > 0; off >>= 1) { if (tid < off) red[tid] += red[tid + off]; __syncthreads(); }
  if (tid == 0) lse[row] = mx + __logf(red[0]);
}

// ---- K7: gather e[t][b][s] = logits[b,s,x[b,t]] - lse[b,s] ------------------
__global__ void k_gather_e(const int* __restrict__ x, const float* __restrict__ logits,
                           const float* __restrict__ lse, float* __restrict__ e) {
  int bt = blockIdx.x; int b = bt / Tv, t = bt % Tv;
  int s = threadIdx.x;
  int g = x[bt];
  int r = b * Sv + s;
  e[(size_t)(t * Bv + b) * Sv + s] = logits[(size_t)r * Gv + g] - lse[r];
}

// ---- K8: sequential HMM forward scan (one wave per (b,m)) -------------------
__global__ void k_scan(const float* __restrict__ logTt, const float* __restrict__ z0row,
                       const float* __restrict__ e, float* __restrict__ zbuf) {
  int wid  = blockIdx.x * 4 + (threadIdx.x >> 5);    // 512 waves
  int lane = threadIdx.x & 31;                       // lane = s2
  int b = wid >> 4, m = wid & 15;
  float tr[32];
#pragma unroll
  for (int s1 = 0; s1 < 32; ++s1) tr[s1] = logTt[(size_t)(m * Sv + lane) * Sv + s1];
  float z = z0row[m * Sv + lane];
  for (int t = 0; t < Tv; ++t) {
    if (t + 8 < Tv) __builtin_prefetch(e + (size_t)((t + 8) * Bv + b) * Sv + lane, 0, 1);
    float et = e[(size_t)(t * Bv + b) * Sv + lane];
    float vmax = -INFINITY;
    float val[32];
#pragma unroll
    for (int s1 = 0; s1 < 32; ++s1) {
      float zs = __shfl(z, s1, 32);
      val[s1] = tr[s1] + zs;
      vmax = fmaxf(vmax, val[s1]);
    }
    float sm = 0.f;
#pragma unroll
    for (int s1 = 0; s1 < 32; ++s1) sm += __expf(val[s1] - vmax);
    z = vmax + __logf(sm) + et;
  }
  zbuf[(size_t)(b * Mv + m) * Sv + lane] = z;
}

// ---- K9: final lse over (M,S) per batch -------------------------------------
__global__ void k_final(const float* __restrict__ zbuf, float* __restrict__ out) {
  int b = blockIdx.x, lane = threadIdx.x;
  float mx = -INFINITY, sm = 0.f;
  for (int i = 0; i < 16; ++i) {
    float v = zbuf[(size_t)b * Mv * Sv + lane + 32 * i] * (1.f / (float)Tv);
    if (v > mx) { sm = sm * __expf(mx - v) + 1.f; mx = v; }
    else          sm += __expf(v - mx);
  }
  for (int off = 16; off > 0; off >>= 1) {
    float omx = __shfl_xor(mx, off, 32);
    float osm = __shfl_xor(sm, off, 32);
    float nm = fmaxf(mx, omx);
    sm = sm * __expf(mx - nm) + osm * __expf(omx - nm);
    mx = nm;
  }
  if (lane == 0) out[b] = mx + __logf(sm);
}

extern "C" void kernel_launch(void* const* d_in, const int* in_sizes, int n_in,
                              void* d_out, int out_size, void* d_ws, size_t ws_size,
                              hipStream_t stream) {
  const int*   x     = (const int*)d_in[0];
  const float* embW  = (const float*)d_in[1];
  const float* vocW  = (const float*)d_in[2];
  const float* vocb  = (const float*)d_in[3];
  const float* initd = (const float*)d_in[4];
  const float* trans = (const float*)d_in[5];
  float* out = (float*)d_out;

  char* ws = (char*)d_ws;
  size_t off = 0;
  auto alloc = [&](size_t bytes) -> void* {
    void* p = ws + off; off += (bytes + 255) & ~(size_t)255; return p;
  };
  float*          xx     = (float*)alloc((size_t)Bv * Tv * Ev * 4);   // 64 MB
  float*          logits = (float*)alloc((size_t)Bv * Sv * Gv * 4);   // 131 MB
  float*          ebuf   = (float*)alloc((size_t)Tv * Bv * Sv * 4);   // 8 MB
  float*          st     = (float*)alloc((size_t)Bv * Sv * Ev * 4);
  unsigned short* stbf   = (unsigned short*)alloc((size_t)Bv * Sv * Ev * 2);
  unsigned short* vwb    = (unsigned short*)alloc((size_t)Gv * Ev * 2); // 16 MB
  float*          amax   = (float*)alloc((size_t)Bv * Sv * 4);
  float*          asum   = (float*)alloc((size_t)Bv * Sv * 4);
  float*          rlse   = (float*)alloc((size_t)Bv * Sv * 4);
  float*          logTt  = (float*)alloc((size_t)Mv * Sv * Sv * 4);
  float*          z0row  = (float*)alloc((size_t)Mv * Sv * 4);
  float*          zbuf   = (float*)alloc((size_t)Bv * Mv * Sv * 4);
  (void)in_sizes; (void)n_in; (void)out_size; (void)ws_size;

  k_prep     <<<dim3(1),            dim3(512), 0, stream>>>(trans, initd, logTt, z0row);
  k_gather   <<<dim3(Bv * Tv),      dim3(64),  0, stream>>>(x, embW, xx);
  k_att_stats<<<dim3(Bv * Sv),      dim3(256), 0, stream>>>(xx, amax, asum);
  k_cvt_vocab<<<dim3((Gv * Ev) / 1024), dim3(256), 0, stream>>>(vocW, vwb, Gv * Ev);
  k_states   <<<dim3(256),          dim3(128), 0, stream>>>(xx, amax, asum, st, stbf);
  k_logits   <<<dim3(2560),         dim3(64),  0, stream>>>(stbf, vwb, vocb, logits);
  k_rowlse   <<<dim3(Bv * Sv),      dim3(256), 0, stream>>>(logits, rlse);
  k_gather_e <<<dim3(Bv * Tv),      dim3(32),  0, stream>>>(x, logits, rlse, ebuf);
  k_scan     <<<dim3(128),          dim3(128), 0, stream>>>(logTt, z0row, ebuf, zbuf);
  k_final    <<<dim3(Bv),           dim3(32),  0, stream>>>(zbuf, out);
}
